// MultiHeadAttentionGPT_5102421148455
// MI455X (gfx1250) — matrix operand order repaired and checked by reading, not compiled, not run on hardware
//
#include <hip/hip_runtime.h>
#include <hip/hip_bf16.h>

#define BB 4
#define TT 2048
#define CC 1024
#define HH 16
#define HS 64

typedef __bf16 bf16;
typedef __attribute__((ext_vector_type(16))) __bf16 v16bf;
typedef __attribute__((ext_vector_type(8)))  __bf16 v8bf;
typedef __attribute__((ext_vector_type(8)))  float  v8f;

// A fragment: lane holds row (lane&15); element j -> K = j + (lane>>4)*16.
static __device__ inline v16bf load_a_frag(const bf16* p0, const bf16* p1) {
    v8bf a = *(const v8bf*)p0;
    v8bf b = *(const v8bf*)p1;
    v16bf r;
#pragma unroll
    for (int i = 0; i < 8; ++i) { r[i] = a[i]; r[i + 8] = b[i]; }
    return r;
}

static __device__ inline v8f wmma_bf16(v16bf a, v16bf b, v8f c) {
    return __builtin_amdgcn_wmma_f32_16x16x32_bf16(false, a, false, b, (short)0, c,
                                                   false, false);
}

// ---------------- precision / layout prep kernels ----------------

__global__ void cast_f32_bf16(const float* __restrict__ in, bf16* __restrict__ out, int n) {
    int i = blockIdx.x * blockDim.x + threadIdx.x;
    int stride = gridDim.x * blockDim.x;
    for (; i < n; i += stride) out[i] = (bf16)in[i];
}

// W[h][c][d] (f32) -> out[h][d][c] (bf16), n = H*HS*C
__global__ void transpose_hcd(const float* __restrict__ w, bf16* __restrict__ out, int n) {
    int i = blockIdx.x * blockDim.x + threadIdx.x;
    int stride = gridDim.x * blockDim.x;
    for (; i < n; i += stride) {
        int c = i % CC;
        int hd = i / CC;
        int d = hd % HS;
        int h = hd / HS;
        out[i] = (bf16)w[((size_t)h * CC + c) * HS + d];
    }
}

// Wp[c][e] (f32) -> out[e][c] (bf16)
__global__ void transpose_sq(const float* __restrict__ w, bf16* __restrict__ out) {
    int i = blockIdx.x * blockDim.x + threadIdx.x;
    int stride = gridDim.x * blockDim.x;
    for (; i < CC * CC; i += stride) {
        int c = i % CC;
        int e = i / CC;
        out[i] = (bf16)w[(size_t)c * CC + e];
    }
}

// ---------------- QKV projection ----------------
// One wave per (b,h,32-row tile) x (q|k|v). 32x64 output tile, software-pipelined K loop.
// __launch_bounds__(32,4): ~256 VGPR budget so the pipelined tile stays in registers.
__global__ void __launch_bounds__(32, 4)
qkv_kernel(const bf16* __restrict__ xb, const bf16* __restrict__ wT,
           bf16* __restrict__ qb, bf16* __restrict__ kb, bf16* __restrict__ vTb) {
    const int lane = threadIdx.x;
    const int n = lane & 15;
    const int g = lane >> 4;
    const int tpb = TT / 32;
    const int tile = blockIdx.x;
    const int proj = blockIdx.y;
    const int b = tile / (HH * tpb);
    const int h = (tile / tpb) % HH;
    const int t0 = (tile % tpb) * 32;

    const bf16* X = xb + (size_t)(b * TT + t0) * CC;
    const bf16* W = wT + (size_t)((proj * HH + h) * HS) * CC;
    const bf16* arow0 = X + (size_t)n * CC;
    const bf16* arow1 = X + (size_t)(16 + n) * CC;

    v8f acc[8];
#pragma unroll
    for (int i = 0; i < 8; ++i) acc[i] = (v8f){};

    auto loadA0 = [&](int c0) { return load_a_frag(arow0 + c0 + g * 16, arow0 + c0 + g * 16 + 8); };
    auto loadA1 = [&](int c0) { return load_a_frag(arow1 + c0 + g * 16, arow1 + c0 + g * 16 + 8); };
    auto loadB  = [&](int c0, int dt) {
        return *(const v16bf*)(W + (size_t)(dt * 16 + n) * CC + c0 + g * 16);
    };

    v16bf a0 = loadA0(0), a1 = loadA1(0);
    v16bf bfr[4];
#pragma unroll
    for (int dt = 0; dt < 4; ++dt) bfr[dt] = loadB(0, dt);

    // unroll 2: ping-pong stages let register renaming delete the stage copies
#pragma unroll 2
    for (int c0 = 0; c0 + 32 < CC; c0 += 32) {
        v16bf na0 = loadA0(c0 + 32);
        v16bf na1 = loadA1(c0 + 32);
        v16bf nb[4];
#pragma unroll
        for (int dt = 0; dt < 4; ++dt) nb[dt] = loadB(c0 + 32, dt);
#pragma unroll
        for (int dt = 0; dt < 4; ++dt) acc[dt] = wmma_bf16(a0, bfr[dt], acc[dt]);
#pragma unroll
        for (int dt = 0; dt < 4; ++dt) acc[4 + dt] = wmma_bf16(a1, bfr[dt], acc[4 + dt]);
        a0 = na0; a1 = na1;
#pragma unroll
        for (int dt = 0; dt < 4; ++dt) bfr[dt] = nb[dt];
    }
#pragma unroll
    for (int dt = 0; dt < 4; ++dt) acc[dt] = wmma_bf16(a0, bfr[dt], acc[dt]);
#pragma unroll
    for (int dt = 0; dt < 4; ++dt) acc[4 + dt] = wmma_bf16(a1, bfr[dt], acc[4 + dt]);

    if (proj < 2) {
        bf16* O = (proj == 0 ? qb : kb) + (size_t)((b * HH + h) * TT + t0) * HS;
#pragma unroll
        for (int dt = 0; dt < 4; ++dt)
#pragma unroll
            for (int r = 0; r < 8; ++r) {
                O[(size_t)(g * 8 + r) * HS + dt * 16 + n] = (bf16)acc[dt][r];
                O[(size_t)(16 + g * 8 + r) * HS + dt * 16 + n] = (bf16)acc[4 + dt][r];
            }
    } else {
        bf16* O = vTb + (size_t)((b * HH + h) * HS) * TT + t0;
#pragma unroll
        for (int dt = 0; dt < 4; ++dt) {
            v8bf pk0, pk1;
#pragma unroll
            for (int r = 0; r < 8; ++r) { pk0[r] = (bf16)acc[dt][r]; pk1[r] = (bf16)acc[4 + dt][r]; }
            *(v8bf*)(O + (size_t)(dt * 16 + n) * TT + g * 8) = pk0;       // t rows 0..15
            *(v8bf*)(O + (size_t)(dt * 16 + n) * TT + 16 + g * 8) = pk1;  // t rows 16..31
        }
    }
}

// ---------------- causal flash attention ----------------
// One wave per (b,h,32-query tile): two 16-row query tiles share each block's K/V fragments.
__global__ void __launch_bounds__(32, 4)
attn_kernel(const bf16* __restrict__ qb, const bf16* __restrict__ kb,
            const bf16* __restrict__ vTb, bf16* __restrict__ attb) {
    const int lane = threadIdx.x;
    const int n = lane & 15;
    const int g = lane >> 4;
    const int tpb = TT / 32;
    const int tile = blockIdx.x;
    const int b = tile / (HH * tpb);
    const int h = (tile / tpb) % HH;
    const int t0 = (tile % tpb) * 32;

    const bf16* Q = qb + (size_t)((b * HH + h) * TT) * HS;
    const bf16* K = kb + (size_t)((b * HH + h) * TT) * HS;
    const bf16* V = vTb + (size_t)((b * HH + h) * HS) * TT;

    __shared__ __align__(32) bf16 pls[2 * 16 * 32];

    v16bf qa[2][2];
#pragma unroll
    for (int tq = 0; tq < 2; ++tq) {
        const bf16* qrow = Q + (size_t)(t0 + tq * 16 + n) * HS;
        qa[tq][0] = load_a_frag(qrow + g * 16, qrow + g * 16 + 8);       // d 0..31
        qa[tq][1] = load_a_frag(qrow + 32 + g * 16, qrow + 40 + g * 16);  // d 32..63
    }

    float mrow[2][8], lrow[2][8];
    v8f acc[2][4];
#pragma unroll
    for (int tq = 0; tq < 2; ++tq) {
#pragma unroll
        for (int r = 0; r < 8; ++r) { mrow[tq][r] = -1e30f; lrow[tq][r] = 0.0f; }
#pragma unroll
        for (int dt = 0; dt < 4; ++dt) acc[tq][dt] = (v8f){};
    }

    const float scale = 0.03125f;  // 1024^-0.5
    const int nkb = t0 / 32 + 1;
    for (int blk = 0; blk < nkb; ++blk) {
        const int u0 = blk * 32;
        if (blk + 1 < nkb) {  // pull next key block toward the WGP (global_prefetch path)
            __builtin_prefetch(K + (size_t)(u0 + 32 + n) * HS + g * 16, 0, 3);
            __builtin_prefetch(K + (size_t)(u0 + 48 + n) * HS + g * 16, 0, 3);
        }
        // shared K fragments for this block
        const bf16* k0 = K + (size_t)(u0 + n) * HS + g * 16;
        const bf16* k1 = K + (size_t)(u0 + 16 + n) * HS + g * 16;
        v16bf b00 = *(const v16bf*)(k0);
        v16bf b01 = *(const v16bf*)(k0 + 32);
        v16bf b10 = *(const v16bf*)(k1);
        v16bf b11 = *(const v16bf*)(k1 + 32);

#pragma unroll
        for (int tq = 0; tq < 2; ++tq) {
            v8f s0 = (v8f){}, s1 = (v8f){};
            s0 = wmma_bf16(qa[tq][0], b00, s0);
            s0 = wmma_bf16(qa[tq][1], b01, s0);
            s1 = wmma_bf16(qa[tq][0], b10, s1);
            s1 = wmma_bf16(qa[tq][1], b11, s1);
            const int tbase = t0 + tq * 16 + g * 8;
            // scale + causal mask (finite sentinel: no inf/NaN in softmax)
#pragma unroll
            for (int r = 0; r < 8; ++r) {
                int t = tbase + r;
                float v0 = s0[r] * scale;
                float v1 = s1[r] * scale;
                if (u0 + n > t) v0 = -1e30f;
                if (u0 + 16 + n > t) v1 = -1e30f;
                s0[r] = v0; s1[r] = v1;
            }
            // block row max across the 16 lanes of each half-wave
            float bm[8];
#pragma unroll
            for (int r = 0; r < 8; ++r) bm[r] = fmaxf(s0[r], s1[r]);
#pragma unroll
            for (int off = 1; off < 16; off <<= 1)
#pragma unroll
                for (int r = 0; r < 8; ++r) bm[r] = fmaxf(bm[r], __shfl_xor(bm[r], off, 32));
            float alpha[8];
#pragma unroll
            for (int r = 0; r < 8; ++r) {
                float mnew = fmaxf(mrow[tq][r], bm[r]);
                alpha[r] = __expf(mrow[tq][r] - mnew);
                mrow[tq][r] = mnew;
            }
#pragma unroll
            for (int r = 0; r < 8; ++r) {
                s0[r] = __expf(s0[r] - mrow[tq][r]);
                s1[r] = __expf(s1[r] - mrow[tq][r]);
            }
            float bs[8];
#pragma unroll
            for (int r = 0; r < 8; ++r) bs[r] = s0[r] + s1[r];
#pragma unroll
            for (int off = 1; off < 16; off <<= 1)
#pragma unroll
                for (int r = 0; r < 8; ++r) bs[r] += __shfl_xor(bs[r], off, 32);
#pragma unroll
            for (int r = 0; r < 8; ++r) lrow[tq][r] = lrow[tq][r] * alpha[r] + bs[r];
#pragma unroll
            for (int dt = 0; dt < 4; ++dt)
#pragma unroll
                for (int r = 0; r < 8; ++r) acc[tq][dt][r] *= alpha[r];
            // C-layout p -> LDS (per-tile slice)
            bf16* pt = pls + tq * 512;
#pragma unroll
            for (int r = 0; r < 8; ++r) {
                pt[(g * 8 + r) * 32 + n] = (bf16)s0[r];
                pt[(g * 8 + r) * 32 + 16 + n] = (bf16)s1[r];
            }
        }
        __syncthreads();
        v16bf pa0 = load_a_frag(&pls[n * 32 + g * 16], &pls[n * 32 + 8 + g * 16]);
        v16bf pa1 = load_a_frag(&pls[512 + n * 32 + g * 16], &pls[512 + n * 32 + 8 + g * 16]);
        // out += p * V ; V fragments shared by both query tiles
#pragma unroll
        for (int dt = 0; dt < 4; ++dt) {
            v16bf vb = *(const v16bf*)(V + (size_t)(dt * 16 + n) * TT + u0 + g * 16);
            acc[0][dt] = wmma_bf16(pa0, vb, acc[0][dt]);
            acc[1][dt] = wmma_bf16(pa1, vb, acc[1][dt]);
        }
        __syncthreads();
    }

    // epilogue: normalize, write concat [b][t][h*HS + d]
#pragma unroll
    for (int tq = 0; tq < 2; ++tq) {
        bf16* O = attb + (size_t)(b * TT + t0 + tq * 16) * CC + h * HS;
#pragma unroll
        for (int r = 0; r < 8; ++r) {
            float inv = 1.0f / lrow[tq][r];
#pragma unroll
            for (int dt = 0; dt < 4; ++dt)
                O[(size_t)(g * 8 + r) * CC + dt * 16 + n] = (bf16)(acc[tq][dt][r] * inv);
        }
    }
}

// ---------------- output projection: out = att @ Wp + bp (f32 out) ----------------
// 32x64 tile per wave, software-pipelined K loop.
__global__ void __launch_bounds__(32, 4)
proj_kernel(const bf16* __restrict__ attb, const bf16* __restrict__ wpT,
            const float* __restrict__ bp, float* __restrict__ out) {
    const int lane = threadIdx.x;
    const int n = lane & 15;
    const int g = lane >> 4;
    const int t0 = blockIdx.x * 32;  // flattened B*T rows
    const int e0 = blockIdx.y * 64;

    const bf16* arow0 = attb + (size_t)(t0 + n) * CC;
    const bf16* arow1 = attb + (size_t)(t0 + 16 + n) * CC;
    const bf16* W = wpT + (size_t)e0 * CC;

    v8f acc[8];
#pragma unroll
    for (int i = 0; i < 8; ++i) acc[i] = (v8f){};

    auto loadA0 = [&](int c0) { return load_a_frag(arow0 + c0 + g * 16, arow0 + c0 + g * 16 + 8); };
    auto loadA1 = [&](int c0) { return load_a_frag(arow1 + c0 + g * 16, arow1 + c0 + g * 16 + 8); };
    auto loadB  = [&](int c0, int dt) {
        return *(const v16bf*)(W + (size_t)(dt * 16 + n) * CC + c0 + g * 16);
    };

    v16bf a0 = loadA0(0), a1 = loadA1(0);
    v16bf bfr[4];
#pragma unroll
    for (int dt = 0; dt < 4; ++dt) bfr[dt] = loadB(0, dt);

#pragma unroll 2
    for (int c0 = 0; c0 + 32 < CC; c0 += 32) {
        v16bf na0 = loadA0(c0 + 32);
        v16bf na1 = loadA1(c0 + 32);
        v16bf nb[4];
#pragma unroll
        for (int dt = 0; dt < 4; ++dt) nb[dt] = loadB(c0 + 32, dt);
#pragma unroll
        for (int dt = 0; dt < 4; ++dt) acc[dt] = wmma_bf16(a0, bfr[dt], acc[dt]);
#pragma unroll
        for (int dt = 0; dt < 4; ++dt) acc[4 + dt] = wmma_bf16(a1, bfr[dt], acc[4 + dt]);
        a0 = na0; a1 = na1;
#pragma unroll
        for (int dt = 0; dt < 4; ++dt) bfr[dt] = nb[dt];
    }
#pragma unroll
    for (int dt = 0; dt < 4; ++dt) acc[dt] = wmma_bf16(a0, bfr[dt], acc[dt]);
#pragma unroll
    for (int dt = 0; dt < 4; ++dt) acc[4 + dt] = wmma_bf16(a1, bfr[dt], acc[4 + dt]);

#pragma unroll
    for (int dt = 0; dt < 4; ++dt) {
        float bias = bp[e0 + dt * 16 + n];
#pragma unroll
        for (int r = 0; r < 8; ++r) {
            out[(size_t)(t0 + g * 8 + r) * CC + e0 + dt * 16 + n] = acc[dt][r] + bias;
            out[(size_t)(t0 + 16 + g * 8 + r) * CC + e0 + dt * 16 + n] = acc[4 + dt][r] + bias;
        }
    }
}

extern "C" void kernel_launch(void* const* d_in, const int* in_sizes, int n_in,
                              void* d_out, int out_size, void* d_ws, size_t ws_size,
                              hipStream_t stream) {
    (void)in_sizes; (void)n_in; (void)out_size; (void)ws_size;
    const float* x  = (const float*)d_in[0];
    const float* Wq = (const float*)d_in[1];
    const float* Wk = (const float*)d_in[2];
    const float* Wv = (const float*)d_in[3];
    const float* Wp = (const float*)d_in[4];
    const float* bp = (const float*)d_in[5];
    float* out = (float*)d_out;

    char* ws = (char*)d_ws;
    size_t off = 0;
    auto alloc = [&](size_t bytes) -> void* {
        void* p = ws + off;
        off += (bytes + 255) & ~(size_t)255;
        return p;
    };
    bf16* xb   = (bf16*)alloc((size_t)BB * TT * CC * 2);       // x in bf16
    bf16* wT   = (bf16*)alloc((size_t)3 * HH * HS * CC * 2);   // Wq/Wk/Wv transposed [p][h][d][c]
    bf16* wpT  = (bf16*)alloc((size_t)CC * CC * 2);            // Wp transposed [e][c]
    bf16* qb   = (bf16*)alloc((size_t)BB * HH * TT * HS * 2);  // [b][h][t][d]
    bf16* kb   = (bf16*)alloc((size_t)BB * HH * TT * HS * 2);  // [b][h][t][d]
    bf16* vTb  = (bf16*)alloc((size_t)BB * HH * HS * TT * 2);  // [b][h][d][t]
    bf16* attb = (bf16*)alloc((size_t)BB * TT * CC * 2);       // concat heads

    cast_f32_bf16<<<4096, 256, 0, stream>>>(x, xb, BB * TT * CC);
    transpose_hcd<<<2048, 256, 0, stream>>>(Wq, wT + (size_t)0 * HH * HS * CC, HH * HS * CC);
    transpose_hcd<<<2048, 256, 0, stream>>>(Wk, wT + (size_t)1 * HH * HS * CC, HH * HS * CC);
    transpose_hcd<<<2048, 256, 0, stream>>>(Wv, wT + (size_t)2 * HH * HS * CC, HH * HS * CC);
    transpose_sq<<<1024, 256, 0, stream>>>(Wp, wpT);

    dim3 gq(BB * HH * (TT / 32), 3);
    qkv_kernel<<<gq, 32, 0, stream>>>(xb, wT, qb, kb, vTb);

    attn_kernel<<<BB * HH * (TT / 32), 32, 0, stream>>>(qb, kb, vTb, attb);

    dim3 gp(BB * TT / 32, CC / 64);
    proj_kernel<<<gp, 32, 0, stream>>>(attb, wpT, bp, out);
}
